// MultiQueryAttention_79199196938375
// MI455X (gfx1250) — matrix operand order repaired and checked by reading, not compiled, not run on hardware
//
#include <hip/hip_runtime.h>
#include <hip/hip_bf16.h>

// ---------------------------------------------------------------------------
// MQA forward for MI455X (gfx1250, wave32, WMMA bf16 16x16x32,
// async-to-LDS staging for K tiles in the attention kernel)
// ---------------------------------------------------------------------------

typedef __attribute__((ext_vector_type(16))) __bf16 v16bf;
typedef __attribute__((ext_vector_type(8)))  float  v8f;

static constexpr int kB  = 2;
static constexpr int kS  = 2048;
static constexpr int kE  = 1024;
static constexpr int kH  = 16;
static constexpr int kHD = 64;
static constexpr int kM  = kB * kS;   // 4096 rows

__device__ __forceinline__ unsigned short f32_to_bf16(float f) {
    union { float f; unsigned int u; } c; c.f = f;
    unsigned int u = c.u;
    u += 0x7FFFu + ((u >> 16) & 1u);   // round to nearest even
    return (unsigned short)(u >> 16);
}

struct Frag {
    union { v16bf v; unsigned short u[16]; uint4 q2[2]; };
};

// A fragment (16x32 bf16, M x K), row-major source with leading dim ld.
// Lane layout (ISA 7.12.2): lane m=L&15, half kh=L>>4:
//   elems 0..7  <-> K = k0 + kh*8 + e
//   elems 8..15 <-> K = k0 + 16 + kh*8 + (e-8)
__device__ __forceinline__ Frag load_a(const unsigned short* base, int ld,
                                       int row0, int k0, int lane) {
    const int m = lane & 15, kh = lane >> 4;
    const unsigned short* p = base + (size_t)(row0 + m) * ld + k0 + kh * 8;
    Frag f;
    f.q2[0] = *(const uint4*)(p);
    f.q2[1] = *(const uint4*)(p + 16);
    return f;
}

// A fragment from a 16x32 bf16 LDS tile (row-major, ld = 32).
__device__ __forceinline__ Frag load_a_lds(const unsigned short* lds, int lane) {
    const int m = lane & 15, kh = lane >> 4;
    const unsigned short* p = lds + m * 32 + kh * 8;
    Frag f;
    f.q2[0] = *(const uint4*)(p);
    f.q2[1] = *(const uint4*)(p + 16);
    return f;
}

// B fragment (32x16 bf16, K x N). Source is column-contiguous-in-K:
// "BT" row-major with leading dim ld, row index = output column.
// Lane n=L&15 holds column col0+n, K = k0 + kh*8 + e (e<8), k0 + 16 + kh*8 + (e-8) (e>=8).
__device__ __forceinline__ Frag load_b(const unsigned short* baseT, int ld,
                                       int col0, int k0, int lane) {
    const int n = lane & 15, kh = lane >> 4;
    const unsigned short* p = baseT + (size_t)(col0 + n) * ld + k0 + kh * 8;
    Frag f;
    f.q2[0] = *(const uint4*)(p);
    f.q2[1] = *(const uint4*)(p + 16);
    return f;
}

// Same as load_b but from an LDS-resident tile.
__device__ __forceinline__ Frag load_b_lds(const unsigned short* lds, int ld,
                                           int col0, int k0, int lane) {
    const int n = lane & 15, kh = lane >> 4;
    const unsigned short* p = lds + (col0 + n) * ld + k0 + kh * 8;
    Frag f;
    f.q2[0] = *(const uint4*)(p);
    f.q2[1] = *(const uint4*)(p + 16);
    return f;
}

__device__ __forceinline__ v8f wmma_bf16(const Frag& a, const Frag& b, v8f c) {
    return __builtin_amdgcn_wmma_f32_16x16x32_bf16(
        /*neg_a=*/false, a.v, /*neg_b=*/false, b.v,
        /*c_mod=*/(short)0, c, /*reuse_a=*/false, /*reuse_b=*/false);
}

__device__ __forceinline__ void wait_asynccnt0() {
#if __has_builtin(__builtin_amdgcn_s_wait_asynccnt)
    __builtin_amdgcn_s_wait_asynccnt(0);
#else
    asm volatile("s_wait_asynccnt 0x0" ::: "memory");
#endif
}

// Async copy of 16B per lane (512B per call) from global memory to LDS.
// GV addressing: VDST = LDS byte offset (wave-relative), VADDR = 64-bit addr.
__device__ __forceinline__ void async_copy_b128(unsigned lds_off,
                                                const void* gaddr) {
    unsigned long long ga = (unsigned long long)(size_t)gaddr;
    asm volatile("global_load_async_to_lds_b128 %0, %1, off"
                 :: "v"(lds_off), "v"(ga) : "memory");
}

// ---------------------------------------------------------------------------
// Elementwise converters
// ---------------------------------------------------------------------------
__global__ __launch_bounds__(256) void convert_f32_bf16_kernel(
        const float* __restrict__ in, unsigned short* __restrict__ out, int n) {
    int i = blockIdx.x * 256 + threadIdx.x;
    if (i < n) out[i] = f32_to_bf16(in[i]);
}

// W [K,N] f32 -> WT [N,K] bf16
__global__ __launch_bounds__(256) void convert_transpose_kernel(
        const float* __restrict__ W, unsigned short* __restrict__ WT,
        int K, int N) {
    int i = blockIdx.x * 256 + threadIdx.x;
    if (i < K * N) {
        int k = i / N, n = i - k * N;
        WT[(size_t)n * K + k] = f32_to_bf16(W[i]);
    }
}

// ---------------------------------------------------------------------------
// Generic bf16 WMMA GEMM: C[M,N] = A[M,K] @ B[K,N] + bias[N]
// B supplied transposed (BT[N,K]). One wave per 32x32 C tile
// (2 A frags + 2 B frags -> 4 WMMAs per k-step: 2x better WMMA:VMEM ratio
//  than one-tile waves, 4x A/B reuse).
// MODE 0: f32 row-major   MODE 1: bf16 row-major
// MODE 2: bf16, output written transposed per batch: out[(b*N+n)*Sdim + s]
// ---------------------------------------------------------------------------
template <int MODE>
__device__ __forceinline__ void store_tile(v8f acc, int mtile, int ntile, int lane,
        const float* __restrict__ bias, float* __restrict__ Cf,
        unsigned short* __restrict__ Ch, int N, int Sdim) {
    const int n = lane & 15, kh = lane >> 4;
    const int col = ntile + n;
    const float bv = bias[col];
#pragma unroll
    for (int r = 0; r < 8; ++r) {
        const int row = mtile + r + 8 * kh;
        const float val = acc[r] + bv;
        if (MODE == 0) {
            Cf[(size_t)row * N + col] = val;
        } else if (MODE == 1) {
            Ch[(size_t)row * N + col] = f32_to_bf16(val);
        } else {
            const int b = row / Sdim, s = row - b * Sdim;
            Ch[((size_t)b * N + col) * Sdim + s] = f32_to_bf16(val);
        }
    }
}

template <int MODE>
__global__ __launch_bounds__(32) void gemm_bias_kernel(
        const unsigned short* __restrict__ A,
        const unsigned short* __restrict__ BT,
        const float* __restrict__ bias,
        float* __restrict__ Cf, unsigned short* __restrict__ Ch,
        int N, int K, int Sdim) {
    const int lane  = threadIdx.x;
    const int mtile = blockIdx.y * 32;
    const int ntile = blockIdx.x * 32;

    v8f acc00 = {}, acc01 = {}, acc10 = {}, acc11 = {};
#pragma unroll 2
    for (int k0 = 0; k0 < K; k0 += 32) {
        Frag a0 = load_a(A, K, mtile,      k0, lane);
        Frag a1 = load_a(A, K, mtile + 16, k0, lane);
        Frag b0 = load_b(BT, K, ntile,      k0, lane);
        Frag b1 = load_b(BT, K, ntile + 16, k0, lane);
        if (k0 + 128 < K) {   // prefetch four k-steps (128 k) ahead (global_prefetch_b8)
            __builtin_prefetch(A  + (size_t)(mtile + (lane & 15)) * K + k0 + 128, 0, 3);
            __builtin_prefetch(BT + (size_t)(ntile + (lane & 15)) * K + k0 + 128, 0, 3);
        }
        acc00 = wmma_bf16(a0, b0, acc00);
        acc01 = wmma_bf16(a0, b1, acc01);
        acc10 = wmma_bf16(a1, b0, acc10);
        acc11 = wmma_bf16(a1, b1, acc11);
    }
    store_tile<MODE>(acc00, mtile,      ntile,      lane, bias, Cf, Ch, N, Sdim);
    store_tile<MODE>(acc01, mtile,      ntile + 16, lane, bias, Cf, Ch, N, Sdim);
    store_tile<MODE>(acc10, mtile + 16, ntile,      lane, bias, Cf, Ch, N, Sdim);
    store_tile<MODE>(acc11, mtile + 16, ntile + 16, lane, bias, Cf, Ch, N, Sdim);
}

// ---------------------------------------------------------------------------
// Flash-style MQA attention.
// One wave per (batch b, head h, 16-query tile). K/V shared across heads.
//   q16    : bf16 [B*S, E]   (head h occupies columns h*64 .. h*64+63)
//   k16    : bf16 [B*S, 64]  (K tile async-staged through LDS per key block)
//   vT16   : bf16 [B, 64, S] (transposed; serves directly as P@V B-fragments)
//   attn16 : bf16 [B*S, E]   output (pre-O-projection)
// ---------------------------------------------------------------------------
__global__ __launch_bounds__(32) void mqa_attn_kernel(
        const unsigned short* __restrict__ q16,
        const unsigned short* __restrict__ k16,
        const unsigned short* __restrict__ vT16,
        unsigned short* __restrict__ attn16) {
    __shared__ unsigned short lds_k[32 * 64];   // K tile: 32 keys x 64 hd (4 KB)
    __shared__ unsigned short lds_p[16 * 32];   // P tile: 16 q x 32 keys (1 KB)

    const int lane = threadIdx.x;
    const int qt = blockIdx.x, h = blockIdx.y, b = blockIdx.z;
    const int n = lane & 15, kh = lane >> 4;
    const int qrow0 = b * kS + qt * 16;

    // Q fragments for this tile: K-dim = hd = 64 -> two 16x32 chunks.
    Frag qf0 = load_a(q16, kE, qrow0, h * kHD +  0, lane);
    Frag qf1 = load_a(q16, kE, qrow0, h * kHD + 32, lane);

    const unsigned short* kbase = k16  + (size_t)b * kS * kHD;
    const unsigned short* vbase = vT16 + (size_t)b * kHD * kS;
    const unsigned lds_k_off = (unsigned)(size_t)(&lds_k[0]);

    v8f o0 = {}, o1 = {}, o2 = {}, o3 = {};
    float mrun[8], lrun[8];
#pragma unroll
    for (int r = 0; r < 8; ++r) { mrun[r] = -3.0e38f; lrun[r] = 0.0f; }

    const float scale = 0.125f;   // 1/sqrt(64)

    for (int j0 = 0; j0 < kS; j0 += 32) {
        // ---- async-stage the 32x64 bf16 K tile (contiguous 4 KB) into LDS.
        // 8 x (32 lanes x 16 B) = 4 KB; tracked by ASYNCcnt.
        {
            const char* gk = (const char*)(kbase + (size_t)j0 * kHD);
#pragma unroll
            for (int i = 0; i < 8; ++i) {
                const unsigned off = (unsigned)(i * 512 + lane * 16);
                async_copy_b128(lds_k_off + off, gk + off);
            }
            wait_asynccnt0();
            __syncthreads();
        }

        // scores: 16 queries x 32 keys (two 16-col WMMA tiles, K=64 each),
        // K-tile B fragments read from LDS (ds_load_b128).
        v8f s0 = {}, s1 = {};
        s0 = wmma_bf16(qf0, load_b_lds(lds_k, kHD,  0,  0, lane), s0);
        s0 = wmma_bf16(qf1, load_b_lds(lds_k, kHD,  0, 32, lane), s0);
        s1 = wmma_bf16(qf0, load_b_lds(lds_k, kHD, 16,  0, lane), s1);
        s1 = wmma_bf16(qf1, load_b_lds(lds_k, kHD, 16, 32, lane), s1);

        // Online softmax update, per accumulator row r (row m = r + 8*kh).
#pragma unroll
        for (int r = 0; r < 8; ++r) {
            float v0 = s0[r] * scale, v1 = s1[r] * scale;
            float mx = fmaxf(v0, v1);
            mx = fmaxf(mx, __shfl_xor(mx, 1, 32));
            mx = fmaxf(mx, __shfl_xor(mx, 2, 32));
            mx = fmaxf(mx, __shfl_xor(mx, 4, 32));
            mx = fmaxf(mx, __shfl_xor(mx, 8, 32));
            const float mnew = fmaxf(mrun[r], mx);
            const float corr = __expf(mrun[r] - mnew);
            const float p0 = __expf(v0 - mnew);
            const float p1 = __expf(v1 - mnew);
            float rs = p0 + p1;
            rs += __shfl_xor(rs, 1, 32);
            rs += __shfl_xor(rs, 2, 32);
            rs += __shfl_xor(rs, 4, 32);
            rs += __shfl_xor(rs, 8, 32);
            lrun[r] = lrun[r] * corr + rs;
            mrun[r] = mnew;
            o0[r] *= corr; o1[r] *= corr; o2[r] *= corr; o3[r] *= corr;
            // stash P (bf16) to LDS, re-laid out C(row m, col key) row-major
            const int m = r + 8 * kh;
            lds_p[m * 32 + n]      = f32_to_bf16(p0);
            lds_p[m * 32 + 16 + n] = f32_to_bf16(p1);
        }
        __syncthreads();

        // P as A fragment (16 q x 32 keys) from LDS; V^T as B fragments
        // loaded directly from global (global_load_b128).
        Frag pa = load_a_lds(lds_p, lane);
        o0 = wmma_bf16(pa, load_b(vbase, kS,  0, j0, lane), o0);
        o1 = wmma_bf16(pa, load_b(vbase, kS, 16, j0, lane), o1);
        o2 = wmma_bf16(pa, load_b(vbase, kS, 32, j0, lane), o2);
        o3 = wmma_bf16(pa, load_b(vbase, kS, 48, j0, lane), o3);
        __syncthreads();
    }

    // Normalize and store (bf16, row-major, head offset h*64).
#pragma unroll
    for (int r = 0; r < 8; ++r) {
        const float inv = 1.0f / lrun[r];
        const int row = qrow0 + r + 8 * kh;
        unsigned short* dst = attn16 + (size_t)row * kE + h * kHD;
        dst[ 0 + n] = f32_to_bf16(o0[r] * inv);
        dst[16 + n] = f32_to_bf16(o1[r] * inv);
        dst[32 + n] = f32_to_bf16(o2[r] * inv);
        dst[48 + n] = f32_to_bf16(o3[r] * inv);
    }
}

// ---------------------------------------------------------------------------
// Launch
// ---------------------------------------------------------------------------
extern "C" void kernel_launch(void* const* d_in, const int* in_sizes, int n_in,
                              void* d_out, int out_size, void* d_ws, size_t ws_size,
                              hipStream_t stream) {
    (void)in_sizes; (void)n_in; (void)out_size; (void)ws_size;

    const float* x  = (const float*)d_in[0];
    const float* wq = (const float*)d_in[1];
    const float* bq = (const float*)d_in[2];
    const float* wk = (const float*)d_in[3];
    const float* bk = (const float*)d_in[4];
    const float* wv = (const float*)d_in[5];
    const float* bv = (const float*)d_in[6];
    const float* wo = (const float*)d_in[7];
    const float* bo = (const float*)d_in[8];
    float* out = (float*)d_out;

    // Workspace layout (bf16 buffers), ~30.7 MB total.
    char* ws = (char*)d_ws;
    size_t off = 0;
    unsigned short* x16    = (unsigned short*)(ws + off); off += (size_t)kM * kE * 2;
    unsigned short* wqT    = (unsigned short*)(ws + off); off += (size_t)kE * kE * 2;
    unsigned short* wkT    = (unsigned short*)(ws + off); off += (size_t)kHD * kE * 2;
    unsigned short* wvT    = (unsigned short*)(ws + off); off += (size_t)kHD * kE * 2;
    unsigned short* woT    = (unsigned short*)(ws + off); off += (size_t)kE * kE * 2;
    unsigned short* q16    = (unsigned short*)(ws + off); off += (size_t)kM * kE * 2;
    unsigned short* k16    = (unsigned short*)(ws + off); off += (size_t)kM * kHD * 2;
    unsigned short* vT16   = (unsigned short*)(ws + off); off += (size_t)kB * kHD * kS * 2;
    unsigned short* attn16 = (unsigned short*)(ws + off); off += (size_t)kM * kE * 2;

    // 1) Convert x -> bf16; weights -> bf16 transposed.
    {
        const int n = kM * kE;
        convert_f32_bf16_kernel<<<(n + 255) / 256, 256, 0, stream>>>(x, x16, n);
    }
    convert_transpose_kernel<<<(kE * kE  + 255) / 256, 256, 0, stream>>>(wq, wqT, kE, kE);
    convert_transpose_kernel<<<(kE * kHD + 255) / 256, 256, 0, stream>>>(wk, wkT, kE, kHD);
    convert_transpose_kernel<<<(kE * kHD + 255) / 256, 256, 0, stream>>>(wv, wvT, kE, kHD);
    convert_transpose_kernel<<<(kE * kE  + 255) / 256, 256, 0, stream>>>(wo, woT, kE, kE);

    // 2) Projections (32x32 tiles per wave).
    {   // q = x @ wq + bq  -> bf16 [4096, 1024]
        dim3 grid(kE / 32, kM / 32);
        gemm_bias_kernel<1><<<grid, 32, 0, stream>>>(x16, wqT, bq, nullptr, q16, kE, kE, kS);
    }
    {   // k = x @ wk + bk  -> bf16 [4096, 64]
        dim3 grid(kHD / 32, kM / 32);
        gemm_bias_kernel<1><<<grid, 32, 0, stream>>>(x16, wkT, bk, nullptr, k16, kHD, kE, kS);
    }
    {   // v = x @ wv + bv  -> bf16 transposed [B, 64, S]
        dim3 grid(kHD / 32, kM / 32);
        gemm_bias_kernel<2><<<grid, 32, 0, stream>>>(x16, wvT, bv, nullptr, vT16, kHD, kE, kS);
    }

    // 3) Attention (flash-style, MQA: K/V shared across heads).
    {
        dim3 grid(kS / 16, kH, kB);
        mqa_attn_kernel<<<grid, 32, 0, stream>>>(q16, k16, vT16, attn16);
    }

    // 4) Output projection: out = attn @ wo + bo  -> f32 [B,S,E]
    {
        dim3 grid(kE / 32, kM / 32);
        gemm_bias_kernel<0><<<grid, 32, 0, stream>>>(attn16, woT, bo, out, nullptr, kE, kE, kS);
    }
}
